// Generator_44538810859675
// MI455X (gfx1250) — matrix operand order repaired and checked by reading, not compiled, not run on hardware
//
#include <hip/hip_runtime.h>
#include <hip/hip_bf16.h>

// ---------------------------------------------------------------------------
// Graph-transformer forward for MI455X (gfx1250, wave32, WMMA).
// Big GEMMs run as bf16 v_wmma_f32_16x16x32_bf16 with the weight tile staged
// in LDS (async global->LDS when available); the 512-wide MLP hidden never
// leaves LDS. fp32 kept for residuals / layernorm statistics.
// ---------------------------------------------------------------------------

typedef __bf16 bf16;
typedef __attribute__((ext_vector_type(16))) __bf16 v16bf;
typedef __attribute__((ext_vector_type(8)))  float  v8f;
typedef int v4i __attribute__((vector_size(16)));

union Frag { v16bf v; uint4 q[2]; };

#if defined(__has_builtin)
#if __has_builtin(__builtin_amdgcn_global_load_async_to_lds_b128) && \
    __has_builtin(__builtin_amdgcn_s_wait_asynccnt)
#define HAVE_ASYNC_LDS 1
#endif
#endif
#ifndef HAVE_ASYNC_LDS
#define HAVE_ASYNC_LDS 0
#endif

#if HAVE_ASYNC_LDS
typedef __attribute__((address_space(1))) v4i* gv4i_p;   // global
typedef __attribute__((address_space(3))) v4i* lv4i_p;   // LDS
#endif

#define WMMA_BF16(a, b, c) \
  __builtin_amdgcn_wmma_f32_16x16x32_bf16(false, (a), false, (b), (short)0, \
                                          (c), false, false)

// ----------------------------- f32 -> bf16 ---------------------------------
__global__ void k_f32_to_bf16(const float* __restrict__ x, bf16* __restrict__ y,
                              size_t n) {
  size_t i  = (size_t)blockIdx.x * blockDim.x + threadIdx.x;
  size_t st = (size_t)gridDim.x * blockDim.x;
  for (; i < n; i += st) y[i] = (bf16)x[i];
}

// --------------------------- generic bf16 GEMM ------------------------------
// C[M,N] = A[M,K] @ W[N,K]^T + bias (+relu) (+resid).
// Block: 128 threads (4 waves). Block tile: 128(M) x 64(N); the 64xK weight
// tile is staged once into LDS (async-to-LDS on gfx1250) and shared by all
// waves. Each wave computes 32x64 (two 16-row subtiles -> 8 WMMA per k-step).
// Requires M%128==0, N%64==0, K%32==0, K<=128.
__global__ void __launch_bounds__(128) k_gemm_bf16(
    const bf16* __restrict__ A, const bf16* __restrict__ W,
    const float* __restrict__ bias, const float* __restrict__ resid,
    float* __restrict__ Cf, bf16* __restrict__ Cbf,
    int M, int N, int K, int relu) {
  const int LW = 136;                      // LDS row stride (272B: conflict pad)
  __shared__ bf16 wlds[64 * LW];           // 17.4 KB
  int mBlocks = M >> 7;
  int bidm = blockIdx.x % mBlocks;
  int nT   = blockIdx.x / mBlocks;
  int wave = threadIdx.x >> 5, lane = threadIdx.x & 31;
  int g = lane >> 4, r = lane & 15;

  // ---- stage W[nT*64 .. +64)[0..K) into LDS ----
  {
    int cpr = K >> 3;                      // 16B chunks per row
    int chunks = 64 * cpr;                 // K=128 -> 1024, K=64 -> 512
    for (int idx = threadIdx.x; idx < chunks; idx += 128) {
      int row = idx / cpr, col = (idx % cpr) << 3;
      const bf16* src = W + (size_t)(nT * 64 + row) * K + col;
      bf16* dst = &wlds[row * LW + col];
#if HAVE_ASYNC_LDS
      __builtin_amdgcn_global_load_async_to_lds_b128((gv4i_p)src, (lv4i_p)dst,
                                                     0, 0);
#else
      *(uint4*)dst = *(const uint4*)src;
#endif
    }
#if HAVE_ASYNC_LDS
    __builtin_amdgcn_s_wait_asynccnt(0);
#endif
    __syncthreads();
  }

  size_t mBase = (size_t)bidm * 128 + (size_t)wave * 32;
  const bf16* Arow0 = A + (mBase + r) * K;
  const bf16* Arow1 = A + (mBase + 16 + r) * K;
  v8f acc[2][4] = {};
  for (int k = 0; k < K; k += 32) {
    Frag a0, a1;
    a0.q[0] = *(const uint4*)(Arow0 + k + 8 * g);
    a0.q[1] = *(const uint4*)(Arow0 + k + 16 + 8 * g);
    a1.q[0] = *(const uint4*)(Arow1 + k + 8 * g);
    a1.q[1] = *(const uint4*)(Arow1 + k + 16 + 8 * g);
#pragma unroll
    for (int t = 0; t < 4; t++) {
      Frag b;
      b.q[0] = *(const uint4*)(&wlds[(t * 16 + r) * LW + k + 8 * g]);
      b.q[1] = *(const uint4*)(&wlds[(t * 16 + r) * LW + k + 16 + 8 * g]);
      acc[0][t] = WMMA_BF16(a0.v, b.v, acc[0][t]);
      acc[1][t] = WMMA_BF16(a1.v, b.v, acc[1][t]);
    }
  }
#pragma unroll
  for (int h = 0; h < 2; h++) {
#pragma unroll
    for (int t = 0; t < 4; t++) {
      int c = nT * 64 + t * 16 + r;
      float bv = bias ? bias[c] : 0.0f;
#pragma unroll
      for (int v = 0; v < 8; v++) {        // D layout: row = v + 8g, col = r
        size_t m = mBase + h * 16 + v + 8 * g;
        float val = acc[h][t][v] + bv;
        if (relu) val = fmaxf(val, 0.0f);
        size_t idx = m * N + c;
        if (resid) val += resid[idx];
        if (Cf)  Cf[idx] = val;
        if (Cbf) Cbf[idx] = (bf16)val;
      }
    }
  }
}

// ----------------------- fused MLP 128 -> 512 -> 128 ------------------------
// 8 waves / block, one 16-row tile.  Hidden (16x512) stays in LDS as bf16.
// out = resid + W1( relu(W0 x + b0) ) + b1
__global__ void __launch_bounds__(256) k_mlp_fused(
    const bf16* __restrict__ X, const bf16* __restrict__ W0,
    const float* __restrict__ b0, const bf16* __restrict__ W1,
    const float* __restrict__ b1, const float* __restrict__ resid,
    float* __restrict__ out, int M) {
  const int LH = 520;                       // padded row stride (16B aligned)
  __shared__ bf16 hid[16 * LH];
  int wave = threadIdx.x >> 5, lane = threadIdx.x & 31;
  int g = lane >> 4, r = lane & 15;
  size_t mBase = (size_t)blockIdx.x * 16;

  // GEMM1: this wave computes hidden cols [wave*64, wave*64+64)
  const bf16* Arow = X + (mBase + r) * 128;
  v8f acc[4] = {};
  for (int k = 0; k < 128; k += 32) {
    Frag a;
    a.q[0] = *(const uint4*)(Arow + k + 8 * g);
    a.q[1] = *(const uint4*)(Arow + k + 16 + 8 * g);
    Frag b[4];
#pragma unroll
    for (int t = 0; t < 4; t++) {
      const bf16* wr = W0 + (size_t)(wave * 64 + t * 16 + r) * 128 + k + 8 * g;
      b[t].q[0] = *(const uint4*)(wr);
      b[t].q[1] = *(const uint4*)(wr + 16);
    }
#pragma unroll
    for (int t = 0; t < 4; t++)
      acc[t] = WMMA_BF16(a.v, b[t].v, acc[t]);
  }
#pragma unroll
  for (int t = 0; t < 4; t++) {
    int c = wave * 64 + t * 16 + r;
    float bv = b0[c];
#pragma unroll
    for (int v = 0; v < 8; v++)
      hid[(v + 8 * g) * LH + c] = (bf16)fmaxf(acc[t][v] + bv, 0.0f);
  }
  __syncthreads();

  // GEMM2: this wave computes out cols [wave*16, wave*16+16), K = 512
  v8f o = {};
  for (int k = 0; k < 512; k += 64) {
    Frag a0, a1, b0f, b1f;
    a0.q[0] = *(const uint4*)(&hid[r * LH + k + 8 * g]);
    a0.q[1] = *(const uint4*)(&hid[r * LH + k + 16 + 8 * g]);
    a1.q[0] = *(const uint4*)(&hid[r * LH + k + 32 + 8 * g]);
    a1.q[1] = *(const uint4*)(&hid[r * LH + k + 48 + 8 * g]);
    const bf16* wr = W1 + (size_t)(wave * 16 + r) * 512 + k + 8 * g;
    b0f.q[0] = *(const uint4*)(wr);
    b0f.q[1] = *(const uint4*)(wr + 16);
    b1f.q[0] = *(const uint4*)(wr + 32);
    b1f.q[1] = *(const uint4*)(wr + 48);
    o = WMMA_BF16(a0.v, b0f.v, o);
    o = WMMA_BF16(a1.v, b1f.v, o);
  }
  int c = wave * 16 + r;
  float bv = b1[c];
#pragma unroll
  for (int v = 0; v < 8; v++) {
    size_t idx = (mBase + v + 8 * g) * 128 + c;
    out[idx] = o[v] + bv + resid[idx];
  }
  (void)M;
}

// ------------------------------ layernorm ----------------------------------
// one wave32 per 128-wide row; 4 elems/lane; shfl_xor reductions.
__global__ void __launch_bounds__(256) k_layernorm(
    const float* __restrict__ x, const float* __restrict__ gw,
    const float* __restrict__ bw, float* __restrict__ yf,
    bf16* __restrict__ ybf, size_t rows) {
  size_t row = (size_t)blockIdx.x * 8 + (threadIdx.x >> 5);
  if (row >= rows) return;
  int lane = threadIdx.x & 31;
  float4 v = ((const float4*)(x + row * 128))[lane];
  float s = v.x + v.y + v.z + v.w;
#pragma unroll
  for (int m = 16; m > 0; m >>= 1) s += __shfl_xor(s, m, 32);
  float mean = s * (1.0f / 128.0f);
  float dx = v.x - mean, dy = v.y - mean, dz = v.z - mean, dw = v.w - mean;
  float q = dx * dx + dy * dy + dz * dz + dw * dw;
#pragma unroll
  for (int m = 16; m > 0; m >>= 1) q += __shfl_xor(q, m, 32);
  float rstd = rsqrtf(q * (1.0f / 128.0f) + 1e-5f);
  int c = lane * 4;
  float o0 = dx * rstd * gw[c + 0] + bw[c + 0];
  float o1 = dy * rstd * gw[c + 1] + bw[c + 1];
  float o2 = dz * rstd * gw[c + 2] + bw[c + 2];
  float o3 = dw * rstd * gw[c + 3] + bw[c + 3];
  if (yf) ((float4*)(yf + row * 128))[lane] = make_float4(o0, o1, o2, o3);
  if (ybf) {
    bf16* p = ybf + row * 128 + c;
    p[0] = (bf16)o0; p[1] = (bf16)o1; p[2] = (bf16)o2; p[3] = (bf16)o3;
  }
}

// ------------------ gated attention elementwise (in-place) ------------------
// a[b,i,j,c] = q[b,i,c]*k[b,j,c]/sqrt(16) * (e+1)*e   (e = linear(edge))
__global__ void k_attn_elem(float* __restrict__ ATT, bf16* __restrict__ ATTbf,
                            const float* __restrict__ Q,
                            const float* __restrict__ Kv, size_t total) {
  size_t i  = (size_t)blockIdx.x * blockDim.x + threadIdx.x;
  size_t st = (size_t)gridDim.x * blockDim.x;
  for (; i < total; i += st) {
    size_t c = i & 127;
    size_t j = (i >> 7) & 127;
    size_t bi = i >> 14;            // b*128 + i
    size_t b = bi >> 7;
    float e = ATT[i];
    float a = Q[bi * 128 + c] * Kv[(b * 128 + j) * 128 + c] * 0.25f *
              (e + 1.0f) * e;
    ATT[i] = a;
    ATTbf[i] = (bf16)a;
  }
}

// ----------------- softmax over j + weighted sum with V ---------------------
__global__ void __launch_bounds__(128) k_softmax_av(
    const float* __restrict__ ATT, const float* __restrict__ V,
    bf16* __restrict__ natt) {
  int bi = blockIdx.x, c = threadIdx.x, b = bi >> 7;
  const float* ar = ATT + ((size_t)bi * 128) * 128 + c;
  const float* vr = V + ((size_t)b * 128) * 128 + c;
  float mx = -3.0e38f;
  for (int j = 0; j < 128; j++) mx = fmaxf(mx, ar[(size_t)j * 128]);
  float s = 0.f, acc = 0.f;
  for (int j = 0; j < 128; j++) {
    float p = __expf(ar[(size_t)j * 128] - mx);
    s += p;
    acc += p * vr[(size_t)j * 128];
  }
  natt[(size_t)bi * 128 + c] = (bf16)(acc / s);
}

// -------------------------- edge symmetrization -----------------------------
__global__ void k_sym(const float* __restrict__ E, float* __restrict__ O,
                      size_t total) {
  size_t i  = (size_t)blockIdx.x * blockDim.x + threadIdx.x;
  size_t st = (size_t)gridDim.x * blockDim.x;
  for (; i < total; i += st) {
    size_t c = i & 127;
    size_t j = (i >> 7) & 127;
    size_t ii = (i >> 14) & 127;
    size_t b = i >> 21;
    size_t tj = (((b * 128 + j) * 128 + ii) * 128) + c;
    O[i] = 0.5f * (E[i] + E[tj]);
  }
}

// --------------------------- node embedding (tiny) --------------------------
__global__ void __launch_bounds__(128) k_embed_node(
    const float* __restrict__ zn, const float* __restrict__ W0,
    const float* __restrict__ b0, const float* __restrict__ W1,
    const float* __restrict__ b1, float* __restrict__ node) {
  __shared__ float hid[64];
  int t = blockIdx.x;
  const float* z = zn + (size_t)t * 16;
  if (threadIdx.x < 64) {
    int o = threadIdx.x;
    float s = b0[o];
#pragma unroll
    for (int k = 0; k < 16; k++) s += z[k] * W0[o * 16 + k];
    hid[o] = fmaxf(s, 0.f);
  }
  __syncthreads();
  int o = threadIdx.x;
  float s = b1[o];
  for (int k = 0; k < 64; k++) s += hid[k] * W1[o * 64 + k];
  node[(size_t)t * 128 + o] = fmaxf(s, 0.f);
}

// ------------------ edge embedding layer 1 (4 -> 64, relu) ------------------
__global__ void k_embed_edge1(const float* __restrict__ ze,
                              const float* __restrict__ W0,
                              const float* __restrict__ b0,
                              bf16* __restrict__ hid, size_t total) {
  size_t i  = (size_t)blockIdx.x * blockDim.x + threadIdx.x;
  size_t st = (size_t)gridDim.x * blockDim.x;
  for (; i < total; i += st) {
    size_t tk = i >> 6;
    int o = (int)(i & 63);
    const float* z = ze + tk * 4;
    const float* w = W0 + o * 4;
    float s = b0[o] + z[0] * w[0] + z[1] * w[1] + z[2] * w[2] + z[3] * w[3];
    hid[i] = (bf16)fmaxf(s, 0.f);
  }
}

// ------------------------- readout linear + softmax -------------------------
__global__ void __launch_bounds__(256) k_readout(
    const float* __restrict__ X, const float* __restrict__ W,
    const float* __restrict__ b, float* __restrict__ out, int Dout,
    size_t rows) {
  size_t row = (size_t)blockIdx.x * blockDim.x + threadIdx.x;
  if (row >= rows) return;
  const float* xr = X + row * 128;
  float vals[16];
  float mx = -3.0e38f;
  for (int o = 0; o < Dout; o++) {
    const float* wr = W + (size_t)o * 128;
    float s = b[o];
    for (int k = 0; k < 128; k++) s += xr[k] * wr[k];
    vals[o] = s;
    mx = fmaxf(mx, s);
  }
  float sum = 0.f;
  for (int o = 0; o < Dout; o++) { vals[o] = __expf(vals[o] - mx); sum += vals[o]; }
  float inv = 1.f / sum;
  for (int o = 0; o < Dout; o++) out[row * Dout + o] = vals[o] * inv;
}

// ===========================================================================
extern "C" void kernel_launch(void* const* d_in, const int* in_sizes, int n_in,
                              void* d_out, int out_size, void* d_ws,
                              size_t ws_size, hipStream_t stream) {
  (void)in_sizes; (void)n_in; (void)out_size; (void)ws_size;
  const size_t NTOK = 1024;                  // B*N
  const size_t ETOK = 131072;                // B*N*N
  const size_t ESZ = ETOK * 128;             // edge tensor elements
  const size_t NSZ = NTOK * 128;

  auto F = [&](int i) { return (const float*)d_in[i]; };

  // bump allocator on workspace
  char* wsp = (char*)d_ws;
  auto alloc = [&](size_t bytes) -> char* {
    char* p = wsp;
    wsp += (bytes + 255) & ~(size_t)255;
    return p;
  };

  float* Eptr = (float*)alloc(ESZ * 4);      // edge state
  float* Aptr = (float*)alloc(ESZ * 4);      // e / attn / mlp-pre scratch
  float* Yptr = (float*)alloc(ESZ * 4);      // y2
  bf16* Ebf = (bf16*)alloc(ESZ * 2);
  bf16* Abf = (bf16*)alloc(ESZ * 2);
  bf16* Ybf = (bf16*)alloc(ESZ * 2);
  bf16* Hemb = (bf16*)alloc(ETOK * 64 * 2);
  float* x1 = (float*)alloc(NSZ * 4);
  bf16* x1bf = (bf16*)alloc(NSZ * 2);
  float* Qb = (float*)alloc(NSZ * 4);
  float* Kb = (float*)alloc(NSZ * 4);
  float* Vb = (float*)alloc(NSZ * 4);
  bf16* nattbf = (bf16*)alloc(NSZ * 2);
  float* X2 = (float*)alloc(NSZ * 4);
  bf16* X2bf = (bf16*)alloc(NSZ * 2);
  float* npre = (float*)alloc(NSZ * 4);
  float* nodeS = (float*)alloc(NSZ * 4);

  auto cvt = [&](const float* s, bf16* dt, size_t n) {
    unsigned bl = (unsigned)((n + 255) / 256);
    if (bl > 32768) bl = 32768;
    k_f32_to_bf16<<<bl, 256, 0, stream>>>(s, dt, n);
  };
  auto toBF = [&](int idx, size_t n) -> bf16* {
    bf16* p = (bf16*)alloc(n * 2);
    cvt(F(idx), p, n);
    return p;
  };
  auto gemm = [&](const bf16* A, const bf16* W, const float* bias,
                  const float* resid, float* Cf, bf16* Cbf, int M, int Nn,
                  int K, int relu) {
    int blocks = (M / 128) * (Nn / 64);
    k_gemm_bf16<<<blocks, 128, 0, stream>>>(A, W, bias, resid, Cf, Cbf, M, Nn,
                                            K, relu);
  };
  auto lnorm = [&](const float* xin, int gi, float* yf, bf16* ybf,
                   size_t rows) {
    k_layernorm<<<(unsigned)((rows + 7) / 8), 256, 0, stream>>>(
        xin, F(gi), F(gi + 1), yf, ybf, rows);
  };

  // ---- weights -> bf16 (input leaf order: z_e, z_n, node_layers,
  //      edge_layers, blocks[ln1,ln3,ln4,ln5,ln6,attn{q,k,v,e,oe,on},mlp,mlp2],
  //      readout_n, readout_e)
  bf16* eW1 = toBF(8, 128 * 64);
  struct BW { bf16 *q, *k, *v, *e, *oe, *on, *m0, *m1, *m20, *m21; };
  BW bw[2];
  for (int p = 0; p < 2; p++) {
    int base = 10 + p * 30;
    bw[p].q = toBF(base + 10, 128 * 128);
    bw[p].k = toBF(base + 12, 128 * 128);
    bw[p].v = toBF(base + 14, 128 * 128);
    bw[p].e = toBF(base + 16, 128 * 128);
    bw[p].oe = toBF(base + 18, 128 * 128);
    bw[p].on = toBF(base + 20, 128 * 128);
    bw[p].m0 = toBF(base + 22, 512 * 128);
    bw[p].m1 = toBF(base + 24, 128 * 512);
    bw[p].m20 = toBF(base + 26, 512 * 128);
    bw[p].m21 = toBF(base + 28, 128 * 512);
  }

  // ---- embeddings
  k_embed_node<<<1024, 128, 0, stream>>>(F(1), F(2), F(3), F(4), F(5), nodeS);
  k_embed_edge1<<<32768, 256, 0, stream>>>(F(0), F(6), F(7), Hemb, ETOK * 64);
  gemm(Hemb, eW1, F(9), nullptr, Eptr, nullptr, (int)ETOK, 128, 64, 1);
  k_sym<<<32768, 256, 0, stream>>>(Eptr, Aptr, ESZ);
  { float* t = Eptr; Eptr = Aptr; Aptr = t; }   // symmetrized edge is state

  // ---- transformer blocks
  for (int p = 0; p < 2; p++) {
    int base = 10 + p * 30;
    lnorm(nodeS, base + 0, x1, x1bf, NTOK);                       // ln1
    gemm(x1bf, bw[p].q, F(base + 11), nullptr, Qb, nullptr, 1024, 128, 128, 0);
    gemm(x1bf, bw[p].k, F(base + 13), nullptr, Kb, nullptr, 1024, 128, 128, 0);
    gemm(x1bf, bw[p].v, F(base + 15), nullptr, Vb, nullptr, 1024, 128, 128, 0);
    cvt(Eptr, Ebf, ESZ);
    gemm(Ebf, bw[p].e, F(base + 17), nullptr, Aptr, nullptr,      // e = lin(edge)
         (int)ETOK, 128, 128, 0);
    k_attn_elem<<<32768, 256, 0, stream>>>(Aptr, Abf, Qb, Kb, ESZ);
    gemm(Abf, bw[p].oe, F(base + 19), Eptr, Yptr, nullptr,        // y2 = out_e + edge
         (int)ETOK, 128, 128, 0);
    k_softmax_av<<<1024, 128, 0, stream>>>(Aptr, Vb, nattbf);
    gemm(nattbf, bw[p].on, F(base + 21), x1, X2, nullptr,         // x2 = x1 + out_n
         1024, 128, 128, 0);
    lnorm(X2, base + 2, X2, X2bf, NTOK);                          // ln3
    lnorm(Yptr, base + 4, Yptr, Ybf, ETOK);                       // ln4
    k_mlp_fused<<<64, 256, 0, stream>>>(X2bf, bw[p].m0, F(base + 23),
                                        bw[p].m1, F(base + 25), X2, npre,
                                        1024);
    lnorm(npre, base + 6, nodeS, nullptr, NTOK);                  // ln5 -> node
    k_mlp_fused<<<8192, 256, 0, stream>>>(Ybf, bw[p].m20, F(base + 27),
                                          bw[p].m21, F(base + 29), Yptr, Aptr,
                                          (int)ETOK);
    lnorm(Aptr, base + 8, Eptr, nullptr, ETOK);                   // ln6 -> edge
  }

  // ---- readouts + output assembly
  float* out = (float*)d_out;
  k_readout<<<4, 256, 0, stream>>>(nodeS, F(70), F(71),
                                   out + 131072 + 16777216, 16, NTOK);
  k_readout<<<512, 256, 0, stream>>>(Eptr, F(72), F(73),
                                     out + 131072 + 16777216 + 16384, 4, ETOK);
  (void)hipMemcpyAsync(out, nodeS, NSZ * 4, hipMemcpyDeviceToDevice, stream);
  (void)hipMemcpyAsync(out + 131072, Eptr, ESZ * 4, hipMemcpyDeviceToDevice,
                       stream);
}
